// Attention_64235530879146
// MI455X (gfx1250) — compile-verified, matrix operand order checked by reading, not run on hardware
//
#include <hip/hip_runtime.h>
#include <hip/hip_bf16.h>

typedef __attribute__((ext_vector_type(16))) __bf16 v16bf;
typedef __attribute__((ext_vector_type(8)))  float  v8f;
typedef __attribute__((ext_vector_type(4)))  unsigned int uint4v;

#define B_   8
#define C_   512
#define T_   1024
#define H_   8
#define CH_  64
#define EPSV 1e-5f
#define QK_SCALE 0.35355339059327379f   // 64^-0.25

union FragBF { v16bf v; uint4v u[2]; };

__device__ __forceinline__ v16bf load_frag(const __hip_bfloat16* p, int e0, int e1) {
    FragBF f;
    f.u[0] = *reinterpret_cast<const uint4v*>(p + e0);
    f.u[1] = *reinterpret_cast<const uint4v*>(p + e1);
    return f.v;
}

// ---------------- Kernel 0: W1 fp32 -> bf16 ----------------
__global__ __launch_bounds__(256) void cvt_w1_kernel(const float* __restrict__ w1,
                                                     __hip_bfloat16* __restrict__ w1b, int n) {
    int i = blockIdx.x * 256 + threadIdx.x;
    if (i < n) w1b[i] = __float2bfloat16(w1[i]);
}

// ---------------- Kernel 1: GroupNorm -> xnT[B,T,C] bf16 ----------------
__global__ __launch_bounds__(256) void gn_kernel(const float* __restrict__ x,
                                                 const float* __restrict__ gw,
                                                 const float* __restrict__ gb,
                                                 __hip_bfloat16* __restrict__ xnT) {
    const int g = blockIdx.x;     // 0..31
    const int b = blockIdx.y;     // 0..7
    const int tid = threadIdx.x;  // 0..255
    __shared__ float s_sum[256];
    __shared__ float s_sq[256];
    const float* xg = x + ((size_t)b * C_ + (size_t)g * 16) * T_;
    float sum = 0.f, sq = 0.f;
#pragma unroll 4
    for (int idx = tid; idx < 16 * T_; idx += 256) {
        float v = xg[idx];
        sum += v; sq += v * v;
    }
    s_sum[tid] = sum; s_sq[tid] = sq;
    __syncthreads();
    for (int s = 128; s > 0; s >>= 1) {
        if (tid < s) { s_sum[tid] += s_sum[tid + s]; s_sq[tid] += s_sq[tid + s]; }
        __syncthreads();
    }
    const float inv_n = 1.0f / (16.0f * T_);
    const float mean  = s_sum[0] * inv_n;
    const float var   = s_sq[0] * inv_n - mean * mean;
    const float rstd  = rsqrtf(var + EPSV);
#pragma unroll 4
    for (int idx = tid; idx < 16 * T_; idx += 256) {
        int cl = idx >> 10;          // channel-in-group
        int t  = idx & 1023;
        int c  = g * 16 + cl;
        float v = (xg[idx] - mean) * rstd * gw[c] + gb[c];
        xnT[((size_t)b * T_ + t) * C_ + c] = __float2bfloat16(v);
    }
}

// ---------------- Kernel 2: QKV GEMM [1536x512]x[512x1024] per batch ----------------
// A 16x16 tile never crosses the q/k/v 512-row boundary nor a 64-row head
// boundary, so `which` and `h` are tile-uniform: branch once, not per element.
__device__ __forceinline__ void store_qkv_tile(const v8f& acc, int Mt, int Nt, int lane, int b,
                                               const float* __restrict__ bias,
                                               __hip_bfloat16* __restrict__ qT,
                                               __hip_bfloat16* __restrict__ kT,
                                               __hip_bfloat16* __restrict__ vv) {
    const int n_local = lane & 15;
    const int mhi = (lane >> 4) * 8;
    const int t = Nt + n_local;
    const int which = Mt >> 9;           // 0=q 1=k 2=v (tile-uniform)
    const int h     = (Mt & 511) >> 6;   // head (tile-uniform)
    const int cbase = (Mt & 63) + mhi;   // channel base for this lane half
    if (which == 0) {
        __hip_bfloat16* dst = qT + (((size_t)b * H_ + h) * T_ + t) * CH_ + cbase;
#pragma unroll
        for (int r = 0; r < 8; ++r)
            dst[r] = __float2bfloat16((acc[r] + bias[Mt + mhi + r]) * QK_SCALE);
    } else if (which == 1) {
        __hip_bfloat16* dst = kT + (((size_t)b * H_ + h) * T_ + t) * CH_ + cbase;
#pragma unroll
        for (int r = 0; r < 8; ++r)
            dst[r] = __float2bfloat16((acc[r] + bias[Mt + mhi + r]) * QK_SCALE);
    } else {
        __hip_bfloat16* dst = vv + (((size_t)b * H_ + h) * CH_ + cbase) * T_ + t;
#pragma unroll
        for (int r = 0; r < 8; ++r)
            dst[(size_t)r * T_] = __float2bfloat16(acc[r] + bias[Mt + mhi + r]);
    }
}

__global__ __launch_bounds__(256) void qkv_kernel(const __hip_bfloat16* __restrict__ w1b, // [1536][512]
                                                  const __hip_bfloat16* __restrict__ xnT, // [B][T][C]
                                                  const float* __restrict__ bias,         // [1536]
                                                  __hip_bfloat16* __restrict__ qT,
                                                  __hip_bfloat16* __restrict__ kT,
                                                  __hip_bfloat16* __restrict__ vv) {
    const int tid  = threadIdx.x;
    const int lane = tid & 31;
    const int wv   = tid >> 5;          // wave 0..7
    const int wm   = wv & 3;            // 4 M sub-blocks of 32
    const int wn   = wv >> 2;           // 2 N sub-blocks of 32
    const int Mbase = blockIdx.x * 128 + wm * 32;
    const int Nbase = blockIdx.y * 64  + wn * 32;
    const int b = blockIdx.z;

    const int lrow = lane & 15;
    const int lk   = (lane >> 4) * 8;   // K half-select per ISA A/B layout

    v8f acc00 = {}, acc01 = {}, acc10 = {}, acc11 = {};
    const __hip_bfloat16* xb = xnT + (size_t)b * T_ * C_;

#pragma unroll 2
    for (int k0 = 0; k0 < C_; k0 += 32) {
        const __hip_bfloat16* a0p = w1b + (size_t)(Mbase + lrow)      * C_ + k0 + lk;
        const __hip_bfloat16* a1p = w1b + (size_t)(Mbase + 16 + lrow) * C_ + k0 + lk;
        v16bf a0 = load_frag(a0p, 0, 16);
        v16bf a1 = load_frag(a1p, 0, 16);
        const __hip_bfloat16* b0p = xb + (size_t)(Nbase + lrow)      * C_ + k0 + lk;
        const __hip_bfloat16* b1p = xb + (size_t)(Nbase + 16 + lrow) * C_ + k0 + lk;
        v16bf bb0 = load_frag(b0p, 0, 16);
        v16bf bb1 = load_frag(b1p, 0, 16);
        acc00 = __builtin_amdgcn_wmma_f32_16x16x32_bf16(false, a0, false, bb0, (short)0, acc00, false, false);
        acc01 = __builtin_amdgcn_wmma_f32_16x16x32_bf16(false, a0, false, bb1, (short)0, acc01, false, false);
        acc10 = __builtin_amdgcn_wmma_f32_16x16x32_bf16(false, a1, false, bb0, (short)0, acc10, false, false);
        acc11 = __builtin_amdgcn_wmma_f32_16x16x32_bf16(false, a1, false, bb1, (short)0, acc11, false, false);
    }
    store_qkv_tile(acc00, Mbase,      Nbase,      lane, b, bias, qT, kT, vv);
    store_qkv_tile(acc01, Mbase,      Nbase + 16, lane, b, bias, qT, kT, vv);
    store_qkv_tile(acc10, Mbase + 16, Nbase,      lane, b, bias, qT, kT, vv);
    store_qkv_tile(acc11, Mbase + 16, Nbase + 16, lane, b, bias, qT, kT, vv);
}

// ---------------- Kernel 3: attention, one block per (b,h,32-row t-tile) ----------------
__global__ __launch_bounds__(256) void attn_kernel(const __hip_bfloat16* __restrict__ qT,
                                                   const __hip_bfloat16* __restrict__ kT,
                                                   const __hip_bfloat16* __restrict__ vv,
                                                   const float* __restrict__ x,
                                                   float* __restrict__ out) {
    extern __shared__ char smem[];
    float*          S      = (float*)smem;                                  // [32][1024] fp32
    __hip_bfloat16* P      = (__hip_bfloat16*)(smem + 32 * 1024 * 4);       // [32][1024] bf16
    float*          red    = (float*)(smem + 32 * 1024 * 4 + 32 * 1024 * 2);// [32][8]
    float*          rowmax = red + 32 * 8;                                  // [32]
    float*          rowsum = rowmax + 32;                                   // [32]

    const int tid  = threadIdx.x;
    const int lane = tid & 31;
    const int wv   = tid >> 5;
    const int t0   = blockIdx.x * 32;
    const int h    = blockIdx.y;
    const int b    = blockIdx.z;

    const __hip_bfloat16* qbh = qT + ((size_t)b * H_ + h) * T_ * CH_;
    const __hip_bfloat16* kbh = kT + ((size_t)b * H_ + h) * T_ * CH_;
    const __hip_bfloat16* vbh = vv + ((size_t)b * H_ + h) * CH_ * T_;

    const int lrow = lane & 15;
    const int lk   = (lane >> 4) * 8;

    // ---- Phase A: S = (Q^T K), 2 row-tiles x 64 col-tiles split over 8 waves ----
    {
        const int rt   = wv & 1;
        const int jgrp = wv >> 1;   // 0..3
        const __hip_bfloat16* ap = qbh + (size_t)(t0 + 16 * rt + lrow) * CH_;
        v16bf aq0 = load_frag(ap, 0 + lk, 16 + lk);    // K = 0..31
        v16bf aq1 = load_frag(ap, 32 + lk, 48 + lk);   // K = 32..63
#pragma unroll 2
        for (int i = 0; i < 16; ++i) {
            int j = jgrp + 4 * i;   // col tile 0..63
            const __hip_bfloat16* bp = kbh + (size_t)(16 * j + lrow) * CH_;
            v16bf bk0 = load_frag(bp, 0 + lk, 16 + lk);
            v16bf bk1 = load_frag(bp, 32 + lk, 48 + lk);
            v8f acc = {};
            acc = __builtin_amdgcn_wmma_f32_16x16x32_bf16(false, aq0, false, bk0, (short)0, acc, false, false);
            acc = __builtin_amdgcn_wmma_f32_16x16x32_bf16(false, aq1, false, bk1, (short)0, acc, false, false);
            int col = 16 * j + lrow;
#pragma unroll
            for (int r = 0; r < 8; ++r) {
                int row = 16 * rt + (lane >> 4) * 8 + r;
                S[row * 1024 + col] = acc[r];
            }
        }
    }
    __syncthreads();

    // ---- Phase B: row max, unnormalized exp -> bf16 P, row sums ----
    {
        const int row  = tid & 31;
        const int part = tid >> 5;  // 8 parts x 128 elements
        const float* Srow = S + row * 1024 + part * 128;
        float m = -3.4e38f;
#pragma unroll 8
        for (int i2 = 0; i2 < 128; ++i2) m = fmaxf(m, Srow[i2]);
        red[row * 8 + part] = m;
        __syncthreads();
        if (tid < 32) {
            float mm = red[tid * 8];
#pragma unroll
            for (int i2 = 1; i2 < 8; ++i2) mm = fmaxf(mm, red[tid * 8 + i2]);
            rowmax[tid] = mm;
        }
        __syncthreads();
        const float rm = rowmax[row];
        float ssum = 0.f;
        __hip_bfloat16* Prow = P + row * 1024 + part * 128;
#pragma unroll 8
        for (int i2 = 0; i2 < 128; ++i2) {
            float e = __expf(Srow[i2] - rm);
            ssum += e;
            Prow[i2] = __float2bfloat16(e);
        }
        red[row * 8 + part] = ssum;
        __syncthreads();
        if (tid < 32) {
            float s2 = 0.f;
#pragma unroll
            for (int i2 = 0; i2 < 8; ++i2) s2 += red[tid * 8 + i2];
            rowsum[tid] = s2;
        }
        __syncthreads();
    }

    // ---- Phase C: O = P x V^T (16x16 tile per wave, K=1024), /rowsum, +residual ----
    {
        const int mt = wv & 1;      // t rows: 16*mt
        const int nt = wv >> 1;     // c cols: 16*nt (0..3)
        v8f acc = {};
        const __hip_bfloat16* Pbase = P   + (size_t)(16 * mt + lrow) * 1024;
        const __hip_bfloat16* vb    = vbh + (size_t)(16 * nt + lrow) * T_;
#pragma unroll 2
        for (int kc = 0; kc < 32; ++kc) {
            int s0 = kc * 32;
            v16bf pa = load_frag(Pbase, s0 + lk, s0 + 16 + lk);
            v16bf vf = load_frag(vb,    s0 + lk, s0 + 16 + lk);
            acc = __builtin_amdgcn_wmma_f32_16x16x32_bf16(false, pa, false, vf, (short)0, acc, false, false);
        }
        const int c = h * CH_ + 16 * nt + (lane & 15);
#pragma unroll
        for (int r = 0; r < 8; ++r) {
            int tl = 16 * mt + (lane >> 4) * 8 + r;
            size_t oidx = ((size_t)b * C_ + c) * T_ + (t0 + tl);
            out[oidx] = acc[r] / rowsum[tl] + x[oidx];
        }
    }
}

extern "C" void kernel_launch(void* const* d_in, const int* in_sizes, int n_in,
                              void* d_out, int out_size, void* d_ws, size_t ws_size,
                              hipStream_t stream) {
    const float* x   = (const float*)d_in[0];
    const float* gw  = (const float*)d_in[1];
    const float* gb  = (const float*)d_in[2];
    const float* w1  = (const float*)d_in[3];
    const float* b1v = (const float*)d_in[4];
    float* out = (float*)d_out;

    char* ws = (char*)d_ws;
    __hip_bfloat16* w1b = (__hip_bfloat16*)ws; ws += (size_t)1536 * 512 * 2;
    __hip_bfloat16* xnT = (__hip_bfloat16*)ws; ws += (size_t)B_ * T_ * C_ * 2;
    __hip_bfloat16* qT  = (__hip_bfloat16*)ws; ws += (size_t)B_ * H_ * T_ * CH_ * 2;
    __hip_bfloat16* kT  = (__hip_bfloat16*)ws; ws += (size_t)B_ * H_ * T_ * CH_ * 2;
    __hip_bfloat16* vv  = (__hip_bfloat16*)ws; ws += (size_t)B_ * H_ * T_ * CH_ * 2;

    cvt_w1_kernel<<<(1536 * 512 + 255) / 256, 256, 0, stream>>>(w1, w1b, 1536 * 512);
    gn_kernel<<<dim3(32, B_), 256, 0, stream>>>(x, gw, gb, xnT);
    qkv_kernel<<<dim3(12, 16, B_), 256, 0, stream>>>(w1b, xnT, b1v, qT, kT, vv);
    size_t smem = (size_t)32 * 1024 * 4 + (size_t)32 * 1024 * 2 + (32 * 8 + 32 + 32) * 4;
    attn_kernel<<<dim3(T_ / 32, H_, B_), 256, smem, stream>>>(qT, kT, vv, x, out);
}
